// MLA_80152679678465
// MI455X (gfx1250) — matrix operand order repaired and checked by reading, not compiled, not run on hardware
//
#include <hip/hip_runtime.h>
#include <cstddef>
#include <cstdint>

// ---------------- problem constants (match reference) ----------------
constexpr int Bq    = 2;
constexpr int Sq    = 2048;
constexpr int DIMc  = 2048;
constexpr int Hc    = 16;
constexpr int DHc   = 128;
constexpr int DRc   = 64;      // d_rope
constexpr int LATc  = 512;
constexpr int BSc   = Bq * Sq;          // 4096 rows
constexpr int NQc   = Hc * (DHc + DRc); // 3072
constexpr int NKVc  = Hc * 2 * DHc;     // 4096
constexpr int DKc   = DHc + DRc;        // 192 combined head dim

typedef __bf16 bf16_t;
typedef __attribute__((ext_vector_type(16))) __bf16 v16bf;
typedef __attribute__((ext_vector_type(8)))  float  v8f;

union FragU { unsigned u[8]; v16bf v; };

// ---------------- CDNA5 async global->LDS staging ----------------
#if __has_builtin(__builtin_amdgcn_global_load_async_to_lds_b128)
#define USE_ASYNC_LDS 1
#else
#define USE_ASYNC_LDS 0
#endif

// Builtin signature (from hipcc diagnostic): param0 = vector_size(16) int __device__*,
// param1 = same in __shared__, then imm offset + imm cpol.
typedef int v4i_gcc __attribute__((vector_size(16)));

__device__ inline void stage_b128(const unsigned* gsrc, unsigned* ldst) {
#if USE_ASYNC_LDS
  __builtin_amdgcn_global_load_async_to_lds_b128(
      (__attribute__((address_space(1))) v4i_gcc*)gsrc,
      (__attribute__((address_space(3))) v4i_gcc*)ldst, 0, 0);
#else
  *(uint4*)ldst = *(const uint4*)gsrc;
#endif
}

template <int N>
__device__ inline void wait_async() {
#if USE_ASYNC_LDS
#if __has_builtin(__builtin_amdgcn_s_wait_asynccnt)
  __builtin_amdgcn_s_wait_asynccnt((unsigned short)N);
#else
  asm volatile("s_wait_asynccnt %0" ::"i"(N) : "memory");
#endif
#endif
}

// ---------------- WMMA helpers ----------------
__device__ inline v8f wmma_bf16(v16bf a, v16bf b, v8f c) {
  // D = A(16x32) * B(32x16) + C, fp32 accumulate
  return __builtin_amdgcn_wmma_f32_16x16x32_bf16(
      /*neg_a=*/false, a, /*neg_b=*/false, b,
      /*c_mod=*/(short)0, c, /*reuse_a=*/false, /*reuse_b=*/false);
}

// A fragment: row-major [row][K] bf16 viewed as u32 (2 bf16/u32), ld in u32.
// Lane layout (ISA 7.12.2): m = lane%16, half = lane/16,
// k = (v/4)*16 + half*8 + (v%4)*2  ->  u32 col = (v/4)*8 + half*4 + (v%4)
__device__ inline v16bf frag_a(const unsigned* p, int row0, int ld) {
  int lane = threadIdx.x & 31, m = lane & 15, hf = lane >> 4;
  FragU f;
#pragma unroll
  for (int v = 0; v < 8; ++v)
    f.u[v] = p[(size_t)(row0 + m) * ld + ((v >> 2) << 3) + (hf << 2) + (v & 3)];
  return f.v;
}

// B fragment from Bt stored row-major [N][K]: n = lane%16,
// k = (v/4)*16 + half*8 + (v%4)*2 -> u32 col = (v/4)*8 + half*4 + (v%4)
__device__ inline v16bf frag_b(const unsigned* p, int n0, int ld) {
  int lane = threadIdx.x & 31, n = lane & 15, hf = lane >> 4;
  FragU f;
#pragma unroll
  for (int v = 0; v < 8; ++v)
    f.u[v] = p[(size_t)(n0 + n) * ld + ((v >> 2) << 3) + (hf << 2) + (v & 3)];
  return f.v;
}

__device__ inline unsigned short f2bf_bits(float x) {
  union { __bf16 h; unsigned short u; } cv;
  cv.h = (__bf16)x;
  return cv.u;
}

// ---------------- elementwise converts ----------------
__global__ void __launch_bounds__(256) k_f32_to_bf16(const float* __restrict__ s,
                                                     bf16_t* __restrict__ d, size_t n) {
  size_t i = (size_t)blockIdx.x * blockDim.x + threadIdx.x;
  if (i < n) d[i] = (bf16_t)s[i];
}

// src f32 [R][C] -> dst bf16 [C][R]  (weights -> Bt layout [N][K])
__global__ void __launch_bounds__(256) k_transpose_to_bf16(const float* __restrict__ s,
                                                           bf16_t* __restrict__ d,
                                                           int R, int C) {
  size_t i = (size_t)blockIdx.x * blockDim.x + threadIdx.x;
  if (i >= (size_t)R * C) return;
  int c = (int)(i % C), r = (int)(i / C);
  d[(size_t)c * R + r] = (bf16_t)s[i];
}

// ---------------- bf16 GEMM: C[M,N] = A[M,K] * Bt[N,K]^T ----------------
// block = 256 threads = 8 waves; block tile 128(M) x 128(N); K step 32.
// Double-buffered LDS, async global->LDS staging (ASYNCcnt pipeline).
// Per wave: 32(M) x 64(N) -> 8 WMMA per K step from 6 fragment gathers.
template <bool OUT_BF16>
__global__ void __launch_bounds__(256) k_gemm_bf16(const bf16_t* __restrict__ A,
                                                   const bf16_t* __restrict__ Bt,
                                                   void* __restrict__ Cout,
                                                   int M, int N, int K) {
  __shared__ unsigned ldsA[2][128 * 16]; // 128 rows x 32 bf16, x2 buffers
  __shared__ unsigned ldsB[2][128 * 16];

  const unsigned* Au = (const unsigned*)A;
  const unsigned* Bu = (const unsigned*)Bt;
  const int tid  = threadIdx.x;
  const int wave = tid >> 5;
  const int wm   = wave & 3;   // 4 waves along M (32 rows each)
  const int wn   = wave >> 2;  // 2 waves along N (64 cols each)
  const int row0 = blockIdx.y * 128;
  const int col0 = blockIdx.x * 128;
  const int Ku   = K >> 1;     // u32 per row

  v8f acc[2][4];
#pragma unroll
  for (int i = 0; i < 2; ++i)
#pragma unroll
    for (int j = 0; j < 4; ++j) acc[i][j] = v8f{};

  // stage one 128x32 A tile + 128x32 B tile: 512+512 uint4, 4 per thread
  auto stage_tile = [&](int t, int buf) {
    const int kk2 = t << 4;  // (t*32)/2 u32 offset
#pragma unroll
    for (int it = 0; it < 2; ++it) {
      int i = tid + (it << 8);
      int r = i >> 2, c4 = i & 3;
      int brow = col0 + r; if (brow >= N) brow = N - 1;  // N=64 case stays in ws
      stage_b128(Au + (size_t)(row0 + r) * Ku + kk2 + (c4 << 2), &ldsA[buf][i << 2]);
      stage_b128(Bu + (size_t)brow * Ku + kk2 + (c4 << 2), &ldsB[buf][i << 2]);
    }
  };

  const int T = K >> 5;
  stage_tile(0, 0);
  for (int t = 0; t < T; ++t) {
    const int cur = t & 1;
    if (t + 1 < T) {
      stage_tile(t + 1, cur ^ 1);
      wait_async<4>();   // 4 in-flight loads belong to tile t+1 -> tile t complete
    } else {
      wait_async<0>();
    }
    __syncthreads();

    v16bf a0 = frag_a(ldsA[cur], wm * 32,      16);
    v16bf a1 = frag_a(ldsA[cur], wm * 32 + 16, 16);
    v16bf b[4];
#pragma unroll
    for (int j = 0; j < 4; ++j) b[j] = frag_b(ldsB[cur], wn * 64 + j * 16, 16);
#pragma unroll
    for (int j = 0; j < 4; ++j) {
      acc[0][j] = wmma_bf16(a0, b[j], acc[0][j]);
      acc[1][j] = wmma_bf16(a1, b[j], acc[1][j]);
    }
    __syncthreads();  // done reading buf before it is overwritten two tiles later
  }

  const int lane = tid & 31, ln = lane & 15, hf = lane >> 4;
#pragma unroll
  for (int i = 0; i < 2; ++i)
#pragma unroll
    for (int j = 0; j < 4; ++j)
#pragma unroll
      for (int r = 0; r < 8; ++r) {
        int row = row0 + wm * 32 + i * 16 + r + 8 * hf;
        int col = col0 + wn * 64 + j * 16 + ln;
        if (col < N) {
          if (OUT_BF16)
            ((bf16_t*)Cout)[(size_t)row * N + col] = (bf16_t)acc[i][j][r];
          else
            ((float*)Cout)[(size_t)row * N + col] = acc[i][j][r];
        }
      }
}

// ---------------- RoPE on q (in place, bf16 [BS][H*192]) ----------------
__global__ void __launch_bounds__(256) k_rope_q(bf16_t* __restrict__ q) {
  int idx = blockIdx.x * blockDim.x + threadIdx.x;
  const int total = BSc * Hc * 32;
  if (idx >= total) return;
  int i  = idx & 31;
  int h  = (idx >> 5) & (Hc - 1);
  int bs = idx >> 9;           // /(32*H)
  int s  = bs & (Sq - 1);
  float inv = __powf(10000.0f, -(float)i / 32.0f);
  float sn, cs;
  __sincosf((float)s * inv, &sn, &cs);
  size_t base = (size_t)bs * NQc + (size_t)h * DKc + DHc;
  float x1 = (float)q[base + i], x2 = (float)q[base + 32 + i];
  q[base + i]      = (bf16_t)(x1 * cs - x2 * sn);
  q[base + 32 + i] = (bf16_t)(x2 * cs + x1 * sn);
}

// --------- assemble Kc [B][H][S][192] and Vt [B][H][128][S] -----------
__global__ void __launch_bounds__(256) k_assemble_kv(const bf16_t* __restrict__ kvb,
                                                     const bf16_t* __restrict__ krb,
                                                     bf16_t* __restrict__ Kc,
                                                     bf16_t* __restrict__ Vt) {
  size_t idx = (size_t)blockIdx.x * blockDim.x + threadIdx.x;
  const size_t total = (size_t)Bq * Hc * Sq * 160;
  if (idx >= total) return;
  int j = (int)(idx % 160);
  int s = (int)((idx / 160) % Sq);
  int h = (int)((idx / (160 * (size_t)Sq)) % Hc);
  int b = (int)(idx / (160 * (size_t)Sq * Hc));
  size_t kcrow = ((size_t)(b * Hc + h) * Sq + s) * DKc;
  if (j < DHc) {
    size_t kvrow = (size_t)(b * Sq + s) * NKVc + (size_t)h * 2 * DHc;
    Kc[kcrow + j] = kvb[kvrow + j];                                       // k_nope
    Vt[((size_t)(b * Hc + h) * DHc + j) * Sq + s] = kvb[kvrow + DHc + j]; // v^T
  } else {
    int i = j - DHc;                      // 0..31, rope pair (i, i+32)
    float inv = __powf(10000.0f, -(float)i / 32.0f);
    float sn, cs;
    __sincosf((float)s * inv, &sn, &cs);
    size_t krow = (size_t)(b * Sq + s) * DRc;
    float x1 = (float)krb[krow + i], x2 = (float)krb[krow + 32 + i];
    Kc[kcrow + DHc + i]      = (bf16_t)(x1 * cs - x2 * sn);
    Kc[kcrow + DHc + 32 + i] = (bf16_t)(x2 * cs + x1 * sn);
  }
}

// ---------------- causal flash attention (one wave per 16 q-rows) -----
// Qc [BS][H*192] bf16, Kc [B][H][S][192] bf16, Vt [B][H][128][S] bf16
// Oa [BS][H*128] bf16
__global__ void __launch_bounds__(128) k_attention(const bf16_t* __restrict__ Qc,
                                                   const bf16_t* __restrict__ Kc,
                                                   const bf16_t* __restrict__ Vt,
                                                   bf16_t* __restrict__ Oa) {
  __shared__ unsigned ldsP[4][16 * 16]; // per-wave 16x32 bf16 patch

  const int tid = threadIdx.x;
  const int wave = tid >> 5, lane = tid & 31, ln = lane & 15, hf = lane >> 4;
  const int gw = blockIdx.x * 4 + wave;
  const int qtiles = Sq >> 4;                 // 128
  const int qt = gw % qtiles;
  const int h  = (gw / qtiles) % Hc;
  const int b  = gw / (qtiles * Hc);
  const int q0 = qt << 4;

  const unsigned* qu = (const unsigned*)Qc;
  const unsigned* ku = (const unsigned*)Kc;
  const unsigned* vu = (const unsigned*)Vt;
  const int ldq = Hc * (DKc >> 1);            // 1536 u32
  const int ldk = DKc >> 1;                   // 96 u32
  const int ldv = Sq >> 1;                    // 1024 u32

  // Q fragments: 6 x (16x32) covering 192 dims, held in registers
  v16bf qf[6];
  const unsigned* qbase = qu + (size_t)(b * Sq + q0) * ldq + h * (DKc >> 1);
#pragma unroll
  for (int i = 0; i < 6; ++i) qf[i] = frag_a(qbase + i * 16, 0, ldq);

  v8f o[8];
#pragma unroll
  for (int d = 0; d < 8; ++d) o[d] = v8f{};
  float mstate[8], lstate[8];
#pragma unroll
  for (int r = 0; r < 8; ++r) { mstate[r] = -1e30f; lstate[r] = 0.0f; }

  const float scale = rsqrtf((float)DKc);
  const unsigned* kbase = ku + (size_t)(b * Hc + h) * Sq * ldk;
  const unsigned* vbase = vu + (size_t)(b * Hc + h) * DHc * ldv;

  for (int kt = 0; kt < q0 + 16; kt += 32) {
    // scores S[16 x 32] = Q . K^T
    v8f s0 = {}, s1 = {};
#pragma unroll
    for (int i = 0; i < 6; ++i) {
      v16bf kb0 = frag_b(kbase + i * 16, kt,      ldk);
      v16bf kb1 = frag_b(kbase + i * 16, kt + 16, ldk);
      s0 = wmma_bf16(qf[i], kb0, s0);
      s1 = wmma_bf16(qf[i], kb1, s1);
    }

    // scale + causal mask + online softmax (rows striped: row = r + 8*hf)
    float p0[8], p1[8], corr[8];
#pragma unroll
    for (int r = 0; r < 8; ++r) {
      int qi  = q0 + r + 8 * hf;
      int k0j = kt + ln, k1j = kt + 16 + ln;
      float v0 = s0[r] * scale; if (k0j > qi) v0 = -1e9f;
      float v1 = s1[r] * scale; if (k1j > qi) v1 = -1e9f;
      float rm = fmaxf(v0, v1);
#pragma unroll
      for (int off = 8; off; off >>= 1) rm = fmaxf(rm, __shfl_xor(rm, off, 16));
      float mnew = fmaxf(mstate[r], rm);
      corr[r] = __expf(mstate[r] - mnew);
      float e0 = __expf(v0 - mnew), e1 = __expf(v1 - mnew);
      float rs = e0 + e1;
#pragma unroll
      for (int off = 8; off; off >>= 1) rs += __shfl_xor(rs, off, 16);
      lstate[r] = lstate[r] * corr[r] + rs;
      mstate[r] = mnew;
      p0[r] = e0; p1[r] = e1;
    }
#pragma unroll
    for (int d = 0; d < 8; ++d)
#pragma unroll
      for (int r = 0; r < 8; ++r) o[d][r] *= corr[r];

    // C-layout P -> A-layout bf16 via wave-private LDS patch
    unsigned short* lps = (unsigned short*)ldsP[wave];
#pragma unroll
    for (int r = 0; r < 8; ++r) {
      int row = r + 8 * hf;
      lps[row * 32 + ln]      = f2bf_bits(p0[r]);
      lps[row * 32 + 16 + ln] = f2bf_bits(p1[r]);
    }
    asm volatile("s_wait_dscnt 0" ::: "memory"); // LDS is in-order per wave; be explicit
    v16bf pa = frag_a(ldsP[wave], 0, 16);

    // O += P(16x32) . V(32x128)
#pragma unroll
    for (int d = 0; d < 8; ++d) {
      v16bf vb = frag_b(vbase + (kt >> 1), d * 16, ldv);
      o[d] = wmma_bf16(pa, vb, o[d]);
    }
  }

  // normalize and write out [BS][H*128]
#pragma unroll
  for (int d = 0; d < 8; ++d)
#pragma unroll
    for (int r = 0; r < 8; ++r) {
      int row = b * Sq + q0 + r + 8 * hf;
      int col = h * DHc + d * 16 + ln;
      Oa[(size_t)row * (Hc * DHc) + col] = (bf16_t)(o[d][r] / lstate[r]);
    }
}

// ---------------- host-side launch ----------------
extern "C" void kernel_launch(void* const* d_in, const int* in_sizes, int n_in,
                              void* d_out, int out_size, void* d_ws, size_t ws_size,
                              hipStream_t stream) {
  (void)in_sizes; (void)n_in; (void)out_size; (void)ws_size;
  const float* x    = (const float*)d_in[0];
  const float* Wq   = (const float*)d_in[1];
  const float* Wdkv = (const float*)d_in[2];
  const float* Wkr  = (const float*)d_in[3];
  const float* Wukv = (const float*)d_in[4];
  const float* Wo   = (const float*)d_in[5];

  char* ws = (char*)d_ws;
  size_t off = 0;
  auto alloc_bf16 = [&](size_t elems) -> bf16_t* {
    bf16_t* p = (bf16_t*)(ws + off);
    off += (elems * sizeof(bf16_t) + 255) & ~(size_t)255;
    return p;
  };
  bf16_t* xb    = alloc_bf16((size_t)BSc * DIMc);   // also reused as attn output
  bf16_t* wqt   = alloc_bf16((size_t)NQc * DIMc);
  bf16_t* wdkvt = alloc_bf16((size_t)LATc * DIMc);
  bf16_t* wkrt  = alloc_bf16((size_t)DRc * DIMc);
  bf16_t* wukvt = alloc_bf16((size_t)NKVc * LATc);
  bf16_t* wot   = alloc_bf16((size_t)DIMc * DIMc);
  bf16_t* qb    = alloc_bf16((size_t)BSc * NQc);
  bf16_t* ckv   = alloc_bf16((size_t)BSc * LATc);
  bf16_t* kvb   = alloc_bf16((size_t)BSc * NKVc);
  bf16_t* krb   = alloc_bf16((size_t)BSc * DRc);
  bf16_t* kc    = alloc_bf16((size_t)Bq * Hc * Sq * DKc);
  bf16_t* vt    = alloc_bf16((size_t)Bq * Hc * DHc * Sq);
  bf16_t* attn  = xb; // alias: xb dead after projection GEMMs

  auto gridN = [](size_t n) { return dim3((unsigned)((n + 255) / 256)); };

  // converts
  k_f32_to_bf16<<<gridN((size_t)BSc * DIMc), 256, 0, stream>>>(x, xb, (size_t)BSc * DIMc);
  k_transpose_to_bf16<<<gridN((size_t)DIMc * NQc),  256, 0, stream>>>(Wq,   wqt,   DIMc, NQc);
  k_transpose_to_bf16<<<gridN((size_t)DIMc * LATc), 256, 0, stream>>>(Wdkv, wdkvt, DIMc, LATc);
  k_transpose_to_bf16<<<gridN((size_t)DIMc * DRc),  256, 0, stream>>>(Wkr,  wkrt,  DIMc, DRc);
  k_transpose_to_bf16<<<gridN((size_t)LATc * NKVc), 256, 0, stream>>>(Wukv, wukvt, LATc, NKVc);
  k_transpose_to_bf16<<<gridN((size_t)DIMc * DIMc), 256, 0, stream>>>(Wo,   wot,   DIMc, DIMc);

  // projection GEMMs (bf16 out), block tile 128x128
  k_gemm_bf16<true><<<dim3((NQc + 127) / 128,  BSc / 128), 256, 0, stream>>>(xb, wqt,   qb,  BSc, NQc,  DIMc);
  k_gemm_bf16<true><<<dim3((LATc + 127) / 128, BSc / 128), 256, 0, stream>>>(xb, wdkvt, ckv, BSc, LATc, DIMc);
  k_gemm_bf16<true><<<dim3((DRc + 127) / 128,  BSc / 128), 256, 0, stream>>>(xb, wkrt,  krb, BSc, DRc,  DIMc);
  k_gemm_bf16<true><<<dim3((NKVc + 127) / 128, BSc / 128), 256, 0, stream>>>(ckv, wukvt, kvb, BSc, NKVc, LATc);

  // RoPE + K/V assembly
  k_rope_q<<<gridN((size_t)BSc * Hc * 32), 256, 0, stream>>>(qb);
  k_assemble_kv<<<gridN((size_t)Bq * Hc * Sq * 160), 256, 0, stream>>>(kvb, krb, kc, vt);

  // attention: 4096 waves, 4 per block
  k_attention<<<dim3((Bq * Hc * (Sq / 16)) / 4), 128, 0, stream>>>(qb, kc, vt, attn);

  // output projection (f32 out)
  k_gemm_bf16<false><<<dim3((DIMc + 127) / 128, BSc / 128), 256, 0, stream>>>(attn, wot, (float*)d_out,
                                                                              BSc, DIMc, DIMc);
}
